// GPTMin_8538394985183
// MI455X (gfx1250) — matrix operand order repaired and checked by reading, not compiled, not run on hardware
//
#include <hip/hip_runtime.h>
#include <hip/hip_bf16.h>
#include <math.h>

// ---------------------------------------------------------------------------
// Model constants (static, from the reference)
// ---------------------------------------------------------------------------
#define VV   50257
#define CTXc 1024
#define DD   768
#define HH   8
#define LL   4
#define BB   2
#define SS   1024
#define BSr  (BB * SS)        // 2048 rows
#define FF   (4 * DD)         // 3072

typedef __attribute__((ext_vector_type(16))) __bf16 v16bf;
typedef __attribute__((ext_vector_type(8)))  __bf16 v8bf;
typedef __attribute__((ext_vector_type(8)))  float  v8f;

// ---------------------------------------------------------------------------
// Tiled GEMM, fp32 in / fp32 out, bf16 WMMA compute, software pipelined.
//   C[M,N] = alpha * A[M,K] x B
//   transB = 1 : B is [N,K] row-major (C = A * B^T), ldb = K stride
//   transB = 0 : B is [K,N] row-major (C = A * B),   ldb = N stride
//
// Invariants relied on by the host code:
//   * M is a multiple of 128 and K a multiple of 32 in every launch.
//   * Only transB=1 launches may have an N edge (logits GEMM, N=50257);
//     transB=0 launches always have N a multiple of 128.
//
// Block: 256 threads (8 wave32). Block tile 128x128, K step 32.
// Waves tiled 4(M) x 2(N); each wave owns 32x64 = 2x4 WMMA fragments.
// ---------------------------------------------------------------------------
#define TM   128
#define TN   128
#define TK   32
#define LDSS 40   // bf16 row stride in LDS: 80 bytes, 16B aligned

// per-thread register staging for one 128x32 fp32 tile (16 floats)
struct TileRegs { float4 v[4]; };

__device__ __forceinline__ void load_a_regs(const float* __restrict__ A,
                                            int lda, int m0, int kk,
                                            int r, int c0, TileRegs& t)
{
    const float* p = A + (size_t)(m0 + r) * lda + (kk + c0);
    t.v[0] = *(const float4*)(p + 0);
    t.v[1] = *(const float4*)(p + 4);
    t.v[2] = *(const float4*)(p + 8);
    t.v[3] = *(const float4*)(p + 12);
}

// transB: B is [N,K]; clamp row address for the N edge, zero after the load.
__device__ __forceinline__ void load_bt_regs(const float* __restrict__ B,
                                             int ldb, int n0, int kk, int N,
                                             int r, int c0, TileRegs& t)
{
    int gn  = n0 + r;
    int gnc = gn < N ? gn : (N - 1);
    const float* p = B + (size_t)gnc * ldb + (kk + c0);
    t.v[0] = *(const float4*)(p + 0);
    t.v[1] = *(const float4*)(p + 4);
    t.v[2] = *(const float4*)(p + 8);
    t.v[3] = *(const float4*)(p + 12);
    if (gn >= N) {
        float4 z = {0.f, 0.f, 0.f, 0.f};
        t.v[0] = z; t.v[1] = z; t.v[2] = z; t.v[3] = z;
    }
}

// !transB: B is [K,N]; coalesced along N, scattered into LDS later (no edge).
__device__ __forceinline__ void load_bn_regs(const float* __restrict__ B,
                                             int ldb, int n0, int kk,
                                             int kidx, int nseg, TileRegs& t)
{
    const float* p = B + (size_t)(kk + kidx) * ldb + (n0 + nseg);
    t.v[0] = *(const float4*)(p + 0);
    t.v[1] = *(const float4*)(p + 4);
    t.v[2] = *(const float4*)(p + 8);
    t.v[3] = *(const float4*)(p + 12);
}

// commit 16 contiguous values: S[r][c0 .. c0+15]  (two 16B ds stores)
__device__ __forceinline__ void commit_row(__bf16* S, int r, int c0,
                                           const TileRegs& t)
{
    const float* f = (const float*)&t;
    v8bf lo, hi;
#pragma unroll
    for (int e = 0; e < 8; ++e) { lo[e] = (__bf16)f[e]; hi[e] = (__bf16)f[8 + e]; }
    *(v8bf*)(&S[r * LDSS + c0])     = lo;
    *(v8bf*)(&S[r * LDSS + c0 + 8]) = hi;
}

// commit 16 n-contiguous values into transposed layout S[n][kidx]
__device__ __forceinline__ void commit_scatter(__bf16* S, int kidx, int nseg,
                                               const TileRegs& t)
{
    const float* f = (const float*)&t;
#pragma unroll
    for (int e = 0; e < 16; ++e)
        S[(nseg + e) * LDSS + kidx] = (__bf16)f[e];
}

__global__ __launch_bounds__(256)
void gemm_bf16_wmma(const float* __restrict__ A, int lda,
                    const float* __restrict__ B, int ldb,
                    float* __restrict__ C, int ldc,
                    int M, int N, int K, float alpha, int transB)
{
    __shared__ __bf16 As[TM * LDSS];
    __shared__ __bf16 Bs[TN * LDSS];

    const int tid  = threadIdx.x;
    const int lane = tid & 31;
    const int wave = tid >> 5;
    const int m0   = blockIdx.y * TM;
    const int n0   = blockIdx.x * TN;
    const int wm   = (wave & 3) * 32;    // wave row offset inside tile
    const int wn   = (wave >> 2) * 64;   // wave col offset inside tile

    const int hl   = lane >> 4;          // half-wave select (0/1)
    const int l16  = lane & 15;

    // tile-fill thread mapping
    const int ar   = tid >> 1;           // row 0..127
    const int ac0  = (tid & 1) * 16;     // col segment {0,16}
    const int kidx = tid >> 3;           // 0..31  (!transB scatter path)
    const int nseg = (tid & 7) * 16;     // 0..112

    v8f zero = {0.f, 0.f, 0.f, 0.f, 0.f, 0.f, 0.f, 0.f};
    v8f acc[2][4];
#pragma unroll
    for (int i = 0; i < 2; ++i)
#pragma unroll
        for (int j = 0; j < 4; ++j) acc[i][j] = zero;

    // ---- prologue: stage K-slice 0 and commit to LDS
    TileRegs ra, rb;
    load_a_regs(A, lda, m0, 0, ar, ac0, ra);
    if (transB) load_bt_regs(B, ldb, n0, 0, N, ar, ac0, rb);
    else        load_bn_regs(B, ldb, n0, 0, kidx, nseg, rb);
    commit_row(As, ar, ac0, ra);
    if (transB) commit_row(Bs, ar, ac0, rb);
    else        commit_scatter(Bs, kidx, nseg, rb);

    for (int kk = 0; kk < K; kk += TK) {
        __syncthreads();

        const bool more = (kk + TK) < K;
        if (more) {   // stage next K-slice while this one computes
            load_a_regs(A, lda, m0, kk + TK, ar, ac0, ra);
            if (transB) load_bt_regs(B, ldb, n0, kk + TK, N, ar, ac0, rb);
            else        load_bn_regs(B, ldb, n0, kk + TK, kidx, nseg, rb);
        }

        // ---- fragments per ISA VGPR layouts (16-byte LDS loads)
        v16bf afrag[2], bfrag[4];
#pragma unroll
        for (int i = 0; i < 2; ++i) {
            const __bf16* ap = &As[(wm + i * 16 + l16) * LDSS];
            v8bf alo = *(const v8bf*)(ap + hl * 8);
            v8bf ahi = *(const v8bf*)(ap + 16 + hl * 8);
            afrag[i] = __builtin_shufflevector(alo, ahi,
                0,1,2,3,4,5,6,7,8,9,10,11,12,13,14,15);
        }
#pragma unroll
        for (int j = 0; j < 4; ++j) {
            const __bf16* bp = &Bs[(wn + j * 16 + l16) * LDSS];
            v8bf blo = *(const v8bf*)(bp + hl * 8);
            v8bf bhi = *(const v8bf*)(bp + 16 + hl * 8);
            bfrag[j] = __builtin_shufflevector(blo, bhi,
                0,1,2,3,4,5,6,7,8,9,10,11,12,13,14,15);
        }

#pragma unroll
        for (int i = 0; i < 2; ++i)
#pragma unroll
            for (int j = 0; j < 4; ++j)
                acc[i][j] = __builtin_amdgcn_wmma_f32_16x16x32_bf16(
                    false, afrag[i], false, bfrag[j],
                    (short)0, acc[i][j], false, false);

        __syncthreads();

        if (more) {   // commit staged slice for the next iteration
            commit_row(As, ar, ac0, ra);
            if (transB) commit_row(Bs, ar, ac0, rb);
            else        commit_scatter(Bs, kidx, nseg, rb);
        }
    }

    // ---- store C (lane L, reg e): row = base + 8*hl + e, col = base + l16
    // M is always a tile multiple; only N needs a guard.
#pragma unroll
    for (int i = 0; i < 2; ++i)
#pragma unroll
        for (int j = 0; j < 4; ++j) {
            int col = n0 + wn + j * 16 + l16;
            if (col < N) {
                int rb_ = m0 + wm + i * 16 + hl * 8;
#pragma unroll
                for (int e = 0; e < 8; ++e)
                    C[(size_t)(rb_ + e) * ldc + col] = alpha * acc[i][j][e];
            }
        }
}

// ---------------------------------------------------------------------------
// Embedding: h[row,:] = wte[x[row],:] + wpe[row % S,:]
// ---------------------------------------------------------------------------
__global__ __launch_bounds__(256)
void embed_kernel(const int* __restrict__ x, const float* __restrict__ wte,
                  const float* __restrict__ wpe, float* __restrict__ h)
{
    int row = blockIdx.x;
    int tok = x[row];
    int pos = row % SS;
    for (int d = threadIdx.x; d < DD; d += 256)
        h[(size_t)row * DD + d] =
            wte[(size_t)tok * DD + d] + wpe[(size_t)pos * DD + d];
}

// ---------------------------------------------------------------------------
// Causal softmax in place on scores [B, S, S]; grid = (S, B)
// ---------------------------------------------------------------------------
__global__ __launch_bounds__(256)
void softmax_causal_kernel(float* __restrict__ sc)
{
    __shared__ float red[256];
    int s = blockIdx.x;
    float* row = sc + ((size_t)blockIdx.y * SS + s) * SS;
    int n = s + 1;                       // valid (unmasked) columns
    int tid = threadIdx.x;

    float m = -3.4e38f;
    for (int t = tid; t < n; t += 256) m = fmaxf(m, row[t]);
    red[tid] = m; __syncthreads();
    for (int w = 128; w > 0; w >>= 1) {
        if (tid < w) red[tid] = fmaxf(red[tid], red[tid + w]);
        __syncthreads();
    }
    m = red[0]; __syncthreads();

    float sum = 0.f;
    for (int t = tid; t < n; t += 256) {
        float e = expf(row[t] - m);
        row[t] = e; sum += e;
    }
    red[tid] = sum; __syncthreads();
    for (int w = 128; w > 0; w >>= 1) {
        if (tid < w) red[tid] += red[tid + w];
        __syncthreads();
    }
    float inv = 1.f / red[0];

    for (int t = tid; t < n; t += 256) row[t] *= inv;
    for (int t = n + tid; t < SS; t += 256) row[t] = 0.f;  // masked -> 0
}

// ---------------------------------------------------------------------------
// LayerNorm (no affine, eps=1e-5), row length D; grid = rows
// ---------------------------------------------------------------------------
__global__ __launch_bounds__(256)
void layernorm_kernel(const float* __restrict__ in, float* __restrict__ out)
{
    __shared__ float red[256];
    size_t row = blockIdx.x;
    const float* r = in + row * DD;
    int tid = threadIdx.x;

    float s = 0.f;
    for (int d = tid; d < DD; d += 256) s += r[d];
    red[tid] = s; __syncthreads();
    for (int w = 128; w > 0; w >>= 1) {
        if (tid < w) red[tid] += red[tid + w];
        __syncthreads();
    }
    float mu = red[0] / (float)DD; __syncthreads();

    float v = 0.f;
    for (int d = tid; d < DD; d += 256) { float t = r[d] - mu; v += t * t; }
    red[tid] = v; __syncthreads();
    for (int w = 128; w > 0; w >>= 1) {
        if (tid < w) red[tid] += red[tid + w];
        __syncthreads();
    }
    float inv = rsqrtf(red[0] / (float)DD + 1e-5f);

    for (int d = tid; d < DD; d += 256)
        out[row * DD + d] = (r[d] - mu) * inv;
}

// ---------------------------------------------------------------------------
// Exact GELU in place: 0.5*x*(1+erf(x/sqrt(2)))
// ---------------------------------------------------------------------------
__global__ __launch_bounds__(256)
void gelu_kernel(float* __restrict__ x, long n)
{
    long i = (long)blockIdx.x * 256 + threadIdx.x;
    if (i < n) {
        float v = x[i];
        x[i] = 0.5f * v * (1.f + erff(v * 0.70710678118654752f));
    }
}

// ---------------------------------------------------------------------------
// dst += src
// ---------------------------------------------------------------------------
__global__ __launch_bounds__(256)
void add_kernel(float* __restrict__ dst, const float* __restrict__ src, long n)
{
    long i = (long)blockIdx.x * 256 + threadIdx.x;
    if (i < n) dst[i] += src[i];
}

// ---------------------------------------------------------------------------
// Host orchestration
// ---------------------------------------------------------------------------
static inline void gemm(hipStream_t st, const float* A, int lda,
                        const float* B, int ldb, float* C, int ldc,
                        int M, int N, int K, float alpha, int transB)
{
    dim3 grid((N + TN - 1) / TN, (M + TM - 1) / TM);
    gemm_bf16_wmma<<<grid, 256, 0, st>>>(A, lda, B, ldb, C, ldc,
                                         M, N, K, alpha, transB);
}

extern "C" void kernel_launch(void* const* d_in, const int* in_sizes, int n_in,
                              void* d_out, int out_size, void* d_ws, size_t ws_size,
                              hipStream_t stream)
{
    (void)in_sizes; (void)n_in; (void)out_size; (void)ws_size;

    const int*   x   = (const int*)d_in[0];
    const float* wte = (const float*)d_in[1];
    const float* wpe = (const float*)d_in[2];
    const float* Wq  = (const float*)d_in[3];
    const float* Wk  = (const float*)d_in[4];
    const float* Wv  = (const float*)d_in[5];
    const float* Wo  = (const float*)d_in[6];
    const float* W1  = (const float*)d_in[7];
    const float* W2  = (const float*)d_in[8];
    float* out = (float*)d_out;

    // Workspace carve-up (~122 MB of fp32)
    float* p  = (float*)d_ws;
    float* h  = p; p += (size_t)BSr * DD;        // [2048, 768]
    float* qh = p; p += (size_t)BSr * DD;        // per-head Q
    float* kh = p; p += (size_t)BSr * DD;        // per-head K
    float* vh = p; p += (size_t)BSr * DD;        // per-head V
    float* sc = p; p += (size_t)BB * SS * SS;    // [2, 1024, 1024]
    float* oc = p; p += (size_t)BSr * (HH * DD); // concat heads [2048, 6144]
    float* yb = p; p += (size_t)BSr * DD;        // attn output y
    float* t1 = p; p += (size_t)BSr * DD;        // LN / ff out
    float* t2 = p; p += (size_t)BSr * FF;        // ff hidden [2048, 3072]

    const float scale = 0.03608439182435161f;    // 1/sqrt(768)
    const long nBD = (long)BSr * DD;
    const long nFF = (long)BSr * FF;

    // h = wte[x] + wpe[pos]
    embed_kernel<<<BSr, 256, 0, stream>>>(x, wte, wpe, h);

    for (int l = 0; l < LL; ++l) {
        for (int hd = 0; hd < HH; ++hd) {
            const float* Wq_lh = Wq + ((size_t)(l * HH + hd)) * DD * DD;
            const float* Wk_lh = Wk + ((size_t)(l * HH + hd)) * DD * DD;
            const float* Wv_lh = Wv + ((size_t)(l * HH + hd)) * DD * DD;

            // q/k/v[b,s,e] = sum_d h[b,s,d] * W[d,e]  (B = [K,N], transB=0)
            gemm(stream, h, DD, Wq_lh, DD, qh, DD, BSr, DD, DD, 1.f, 0);
            gemm(stream, h, DD, Wk_lh, DD, kh, DD, BSr, DD, DD, 1.f, 0);
            gemm(stream, h, DD, Wv_lh, DD, vh, DD, BSr, DD, DD, 1.f, 0);

            // scores[b] = (q_b @ k_b^T) * scale   (transB=1)
            for (int b = 0; b < BB; ++b)
                gemm(stream, qh + (size_t)b * SS * DD, DD,
                             kh + (size_t)b * SS * DD, DD,
                             sc + (size_t)b * SS * SS, SS,
                             SS, SS, DD, scale, 1);

            // causal softmax
            softmax_causal_kernel<<<dim3(SS, BB), 256, 0, stream>>>(sc);

            // o[b] = attn_b @ v_b  -> concat buffer columns [hd*768, ...)
            for (int b = 0; b < BB; ++b)
                gemm(stream, sc + (size_t)b * SS * SS, SS,
                             vh + (size_t)b * SS * DD, DD,
                             oc + (size_t)b * SS * (HH * DD) + (size_t)hd * DD,
                             HH * DD,
                             SS, DD, SS, 1.f, 0);
        }

        // y = o_concat @ Wo^T   (Wo[l] is [768, 6144], transB=1)
        gemm(stream, oc, HH * DD, Wo + (size_t)l * DD * HH * DD, HH * DD,
             yb, DD, BSr, DD, HH * DD, 1.f, 1);

        // ff = GELU(LN(y) @ W1^T) @ W2^T
        layernorm_kernel<<<BSr, 256, 0, stream>>>(yb, t1);
        gemm(stream, t1, DD, W1 + (size_t)l * FF * DD, DD,
             t2, FF, BSr, FF, DD, 1.f, 1);
        gelu_kernel<<<(nFF + 255) / 256, 256, 0, stream>>>(t2, nFF);
        gemm(stream, t2, FF, W2 + (size_t)l * DD * FF, FF,
             t1, DD, BSr, DD, FF, 1.f, 1);

        // y += ff ; h += y
        add_kernel<<<(nBD + 255) / 256, 256, 0, stream>>>(yb, t1, nBD);
        add_kernel<<<(nBD + 255) / 256, 256, 0, stream>>>(h, yb, nBD);
    }

    // logits = h @ wte^T  (tied weights, N = 50257 with edge guards)
    gemm(stream, h, DD, wte, DD, out, VV, BSr, VV, DD, 1.f, 1);
}
